// GNN_classification_56642028700255
// MI455X (gfx1250) — matrix operand order repaired and checked by reading, not compiled, not run on hardware
//
#include <hip/hip_runtime.h>
#include <hip/hip_bf16.h>

#define NN      100000
#define NE      3200000
#define DDIM    256
#define NT      8192
#define OUTD    10
#define LGNN    4
#define LLIN    2
#define EPSF    1e-5f

typedef __attribute__((ext_vector_type(16))) _Float16 v16h;
typedef __attribute__((ext_vector_type(8)))  _Float16 h8;
typedef __attribute__((ext_vector_type(4)))  _Float16 h4;
typedef __attribute__((ext_vector_type(8)))  float    v8f;
typedef __attribute__((ext_vector_type(4)))  float    f4;

// ---------------- weight transpose + f16 cast: wt[l][n][k] = W[l][k][n] ----------------
__global__ void wcast_kernel(const float* __restrict__ gnnW, const float* __restrict__ linW,
                             _Float16* __restrict__ wt) {
  int idx = blockIdx.x * blockDim.x + threadIdx.x;
  if (idx >= (LGNN + LLIN) * DDIM * DDIM) return;
  int l = idx / (DDIM * DDIM);
  int r = idx % (DDIM * DDIM);
  int n = r / DDIM, k = r % DDIM;
  float w = (l < LGNN) ? gnnW[(size_t)l * DDIM * DDIM + k * DDIM + n]
                       : linW[(size_t)(l - LGNN) * DDIM * DDIM + k * DDIM + n];
  wt[idx] = (_Float16)w;
}

// ---------------- embedding gather -> fp32 + f16 ----------------
__global__ void embed_kernel(const int* __restrict__ tokens, const float* __restrict__ emb,
                             float* __restrict__ x32, _Float16* __restrict__ x16) {
  long long idx = (long long)blockIdx.x * blockDim.x + threadIdx.x;
  int n = (int)(idx >> 6);
  if (n >= NN) return;
  int d = ((int)idx & 63) * 4;
  f4 v = *(const f4*)(emb + (size_t)tokens[n] * DDIM + d);
  *(f4*)(x32 + (size_t)n * DDIM + d) = v;
  h4 hv; hv[0] = (_Float16)v[0]; hv[1] = (_Float16)v[1];
  hv[2] = (_Float16)v[2]; hv[3] = (_Float16)v[3];
  *(h4*)(x16 + (size_t)n * DDIM + d) = hv;
}

// ---------------- degree: init (self loop), count, rsqrt ----------------
__global__ void deg_init_kernel(float* __restrict__ dinv) {
  int n = blockIdx.x * blockDim.x + threadIdx.x;
  if (n < NN) dinv[n] = 1.0f;
}
__global__ void deg_count_kernel(const int* __restrict__ dst, float* __restrict__ dinv) {
  int e = blockIdx.x * blockDim.x + threadIdx.x;
  if (e >= NE) return;
  __hip_atomic_fetch_add(&dinv[dst[e]], 1.0f, __ATOMIC_RELAXED, __HIP_MEMORY_SCOPE_AGENT);
}
__global__ void deg_finish_kernel(float* __restrict__ dinv) {
  int n = blockIdx.x * blockDim.x + threadIdx.x;
  if (n < NN) dinv[n] = rsqrtf(dinv[n]);
}

// ---------------- WMMA GEMM: H[nrows,256] = Xh[nrows,256] @ Wt^T ----------------
// block = 256 threads = 8 waves; wave w computes rows [blk*128 + w*16, +16) x all 256 cols.
// Per K-step the 256x32 f16 B-tile (16 KB) is staged in LDS cooperatively so the
// WMMA operands come from ds_load_b128 instead of long-latency global loads.
__global__ void __launch_bounds__(256) gemm_wmma_kernel(const _Float16* __restrict__ Xh,
                                                        const _Float16* __restrict__ Wt,
                                                        float* __restrict__ H, int nrows) {
  __shared__ _Float16 ldsB[DDIM * 32];  // [n][kk], kk = K within the step
  int tid  = threadIdx.x;
  int wave = tid >> 5;
  int lane = tid & 31;
  int mbase = blockIdx.x * 128 + wave * 16;
  int lm = lane & 15;                   // M (A/C) or N (B/C) sub-index
  int lh = lane >> 4;                   // half-wave select
  int m = mbase + lm;
  bool mok = (m < nrows);
  const _Float16* arow = Xh + (size_t)m * DDIM;

  v8f acc[16];
  v8f zero8 = {0.f, 0.f, 0.f, 0.f, 0.f, 0.f, 0.f, 0.f};
#pragma unroll
  for (int t = 0; t < 16; ++t) acc[t] = zero8;

  for (int ks = 0; ks < 8; ++ks) {      // K in steps of 32
    int k0 = ks * 32;
    // cooperative B-tile fill: thread tid copies row n=tid (64 B)
    __syncthreads();
    {
      const f4* srcp = (const f4*)(Wt + (size_t)tid * DDIM + k0);
      f4* dstp = (f4*)(ldsB + tid * 32);
      dstp[0] = srcp[0]; dstp[1] = srcp[1]; dstp[2] = srcp[2]; dstp[3] = srcp[3];
    }
    __syncthreads();

    v16h a;
    if (mok) {
      h8 alo = *(const h8*)(arow + k0 + lh * 8);        // K = kbase..kbase+7
      h8 ahi = *(const h8*)(arow + k0 + 16 + lh * 8);   // K = 16+kbase..
#pragma unroll
      for (int j = 0; j < 8; ++j) { a[j] = alo[j]; a[8 + j] = ahi[j]; }
    } else {
#pragma unroll
      for (int j = 0; j < 16; ++j) a[j] = (_Float16)0.0f;
    }
#pragma unroll
    for (int nt = 0; nt < 16; ++nt) {
      const _Float16* wl = ldsB + (nt * 16 + lm) * 32 + lh * 8;
      h8 blo = *(const h8*)(wl);
      h8 bhi = *(const h8*)(wl + 16);
      v16h b;
#pragma unroll
      for (int j = 0; j < 8; ++j) { b[j] = blo[j]; b[8 + j] = bhi[j]; }
      acc[nt] = __builtin_amdgcn_wmma_f32_16x16x32_f16(
          false, a, false, b, (short)0, acc[nt], false, false);
    }
  }
#pragma unroll
  for (int nt = 0; nt < 16; ++nt) {
    int col = nt * 16 + lm;
#pragma unroll
    for (int j = 0; j < 8; ++j) {
      int row = mbase + j + 8 * lh;
      if (row < nrows) H[(size_t)row * DDIM + col] = acc[nt][j];
    }
  }
}

// ---------------- y init with self-loop: y[n] = dinv[n]^2 * h[n] ----------------
__global__ void yinit_kernel(const float* __restrict__ H, const float* __restrict__ dinv,
                             float* __restrict__ Y) {
  long long idx = (long long)blockIdx.x * blockDim.x + threadIdx.x;
  int n = (int)(idx >> 6);
  if (n >= NN) return;
  int d = ((int)idx & 63) * 4;
  float w = dinv[n]; w = w * w;
  f4 hv = *(const f4*)(H + (size_t)n * DDIM + d);
  f4 o; o[0] = w * hv[0]; o[1] = w * hv[1]; o[2] = w * hv[2]; o[3] = w * hv[3];
  *(f4*)(Y + (size_t)n * DDIM + d) = o;
}

// ---------------- edge scatter: y[dst] += dinv[src]*dinv[dst]*h[src] ----------------
__global__ void scatter_kernel(const int* __restrict__ src, const int* __restrict__ dst,
                               const float* __restrict__ dinv, const float* __restrict__ H,
                               float* __restrict__ Y) {
  long long idx = (long long)blockIdx.x * blockDim.x + threadIdx.x;
  int e = (int)(idx >> 6);
  if (e >= NE) return;
  int d = ((int)idx & 63) * 4;
  int s = src[e], t = dst[e];
  float w = dinv[s] * dinv[t];
  f4 hv = *(const f4*)(H + (size_t)s * DDIM + d);
  float* yp = Y + (size_t)t * DDIM + d;
  __hip_atomic_fetch_add(yp + 0, w * hv[0], __ATOMIC_RELAXED, __HIP_MEMORY_SCOPE_AGENT);
  __hip_atomic_fetch_add(yp + 1, w * hv[1], __ATOMIC_RELAXED, __HIP_MEMORY_SCOPE_AGENT);
  __hip_atomic_fetch_add(yp + 2, w * hv[2], __ATOMIC_RELAXED, __HIP_MEMORY_SCOPE_AGENT);
  __hip_atomic_fetch_add(yp + 3, w * hv[3], __ATOMIC_RELAXED, __HIP_MEMORY_SCOPE_AGENT);
}

// ---------------- bias + LayerNorm + ReLU; writes fp32 + f16 ----------------
// one wave (32 lanes) per row; 8 elements/lane; wave32 shuffle reduction.
__global__ void ln_kernel(const float* __restrict__ Y, const float* __restrict__ bias,
                          const float* __restrict__ gamma, const float* __restrict__ beta,
                          float* __restrict__ X32, _Float16* __restrict__ X16, int nrows) {
  int row = blockIdx.x * 8 + (threadIdx.x >> 5);
  int lane = threadIdx.x & 31;
  if (row >= nrows) return;                   // wave-uniform
  int dbase = lane * 8;
  const float* yp = Y + (size_t)row * DDIM + dbase;
  f4 a = *(const f4*)(yp);
  f4 bsec = *(const f4*)(yp + 4);
  float v[8];
#pragma unroll
  for (int j = 0; j < 4; ++j) {
    v[j] = a[j] + bias[dbase + j];
    v[4 + j] = bsec[j] + bias[dbase + 4 + j];
  }
  float s = 0.f, s2 = 0.f;
#pragma unroll
  for (int j = 0; j < 8; ++j) { s += v[j]; s2 += v[j] * v[j]; }
#pragma unroll
  for (int off = 16; off > 0; off >>= 1) {
    s += __shfl_xor(s, off, 32);
    s2 += __shfl_xor(s2, off, 32);
  }
  float mu = s * (1.0f / DDIM);
  float var = s2 * (1.0f / DDIM) - mu * mu;
  float rs = rsqrtf(var + EPSF);
  f4 o0, o1; h8 oh;
#pragma unroll
  for (int j = 0; j < 8; ++j) {
    float z = (v[j] - mu) * rs * gamma[dbase + j] + beta[dbase + j];
    z = fmaxf(z, 0.0f);
    if (j < 4) o0[j] = z; else o1[j - 4] = z;
    oh[j] = (_Float16)z;
  }
  *(f4*)(X32 + (size_t)row * DDIM + dbase) = o0;
  *(f4*)(X32 + (size_t)row * DDIM + dbase + 4) = o1;
  *(h8*)(X16 + (size_t)row * DDIM + dbase) = oh;
}

// ---------------- target gather -> f16 for lin GEMM ----------------
__global__ void gather_kernel(const int* __restrict__ tgt, const float* __restrict__ X32,
                              _Float16* __restrict__ G16) {
  long long idx = (long long)blockIdx.x * blockDim.x + threadIdx.x;
  int t = (int)(idx >> 6);
  if (t >= NT) return;
  int d = ((int)idx & 63) * 4;
  f4 v = *(const f4*)(X32 + (size_t)tgt[t] * DDIM + d);
  h4 hv; hv[0] = (_Float16)v[0]; hv[1] = (_Float16)v[1];
  hv[2] = (_Float16)v[2]; hv[3] = (_Float16)v[3];
  *(h4*)(G16 + (size_t)t * DDIM + d) = hv;
}

// ---------------- final projection: out = Xg @ out_W + out_b ----------------
__global__ void outproj_kernel(const float* __restrict__ Xg, const float* __restrict__ outW,
                               const float* __restrict__ outB, float* __restrict__ out) {
  int idx = blockIdx.x * blockDim.x + threadIdx.x;
  if (idx >= NT * OUTD) return;
  int t = idx / OUTD, o = idx % OUTD;
  const float* xr = Xg + (size_t)t * DDIM;
  float acc = outB[o];
#pragma unroll 4
  for (int k = 0; k < DDIM; ++k) acc += xr[k] * outW[k * OUTD + o];
  out[idx] = acc;
}

extern "C" void kernel_launch(void* const* d_in, const int* in_sizes, int n_in,
                              void* d_out, int out_size, void* d_ws, size_t ws_size,
                              hipStream_t stream) {
  const int*   tokens = (const int*)d_in[0];
  const int*   edges  = (const int*)d_in[1];
  const int*   tgt    = (const int*)d_in[2];
  const float* emb    = (const float*)d_in[3];
  const float* gnnW   = (const float*)d_in[4];
  const float* gnnB   = (const float*)d_in[5];
  const float* gnnG   = (const float*)d_in[6];
  const float* gnnBe  = (const float*)d_in[7];
  const float* linW   = (const float*)d_in[8];
  const float* linB   = (const float*)d_in[9];
  const float* linG   = (const float*)d_in[10];
  const float* linBe  = (const float*)d_in[11];
  const float* outW   = (const float*)d_in[12];
  const float* outB   = (const float*)d_in[13];
  float* out = (float*)d_out;

  const int* srcE = edges;
  const int* dstE = edges + NE;

  char* wsb = (char*)d_ws;
  size_t off = 0;
  auto carve = [&](size_t bytes) -> void* {
    void* p = wsb + off;
    off += (bytes + 255) & ~(size_t)255;
    return p;
  };
  float*    x32  = (float*)   carve((size_t)NN * DDIM * 4);
  _Float16* x16  = (_Float16*)carve((size_t)NN * DDIM * 2);
  float*    hbuf = (float*)   carve((size_t)NN * DDIM * 4);
  float*    ybuf = (float*)   carve((size_t)NN * DDIM * 4);
  float*    dinv = (float*)   carve((size_t)NN * 4);
  _Float16* wt16 = (_Float16*)carve((size_t)(LGNN + LLIN) * DDIM * DDIM * 2);
  // target-stage buffers alias the (no longer needed) h/y node buffers
  _Float16* g16  = (_Float16*)hbuf;
  float*    g32  = (float*)((char*)hbuf + (((size_t)NT * DDIM * 2 + 255) & ~(size_t)255));
  float*    yg   = ybuf;

  // 1. weight convert
  {
    int n = (LGNN + LLIN) * DDIM * DDIM;
    wcast_kernel<<<(n + 255) / 256, 256, 0, stream>>>(gnnW, linW, wt16);
  }
  // 2. embedding
  embed_kernel<<<(NN * 64 + 255) / 256, 256, 0, stream>>>(tokens, emb, x32, x16);
  // 3. degrees (once; edges fixed)
  deg_init_kernel<<<(NN + 255) / 256, 256, 0, stream>>>(dinv);
  deg_count_kernel<<<(NE + 255) / 256, 256, 0, stream>>>(dstE, dinv);
  deg_finish_kernel<<<(NN + 255) / 256, 256, 0, stream>>>(dinv);
  // 4. GNN layers
  for (int l = 0; l < LGNN; ++l) {
    gemm_wmma_kernel<<<(NN + 127) / 128, 256, 0, stream>>>(
        x16, wt16 + (size_t)l * DDIM * DDIM, hbuf, NN);
    yinit_kernel<<<(NN * 64 + 255) / 256, 256, 0, stream>>>(hbuf, dinv, ybuf);
    scatter_kernel<<<(int)(((long long)NE * 64 + 255) / 256), 256, 0, stream>>>(
        srcE, dstE, dinv, hbuf, ybuf);
    ln_kernel<<<(NN + 7) / 8, 256, 0, stream>>>(
        ybuf, gnnB + (size_t)l * DDIM, gnnG + (size_t)l * DDIM, gnnBe + (size_t)l * DDIM,
        x32, x16, NN);
  }
  // 5. gather targets (f16 for GEMM)
  gather_kernel<<<(NT * 64 + 255) / 256, 256, 0, stream>>>(tgt, x32, g16);
  // 6. linear layers on 8192 targets
  for (int l = 0; l < LLIN; ++l) {
    gemm_wmma_kernel<<<(NT + 127) / 128, 256, 0, stream>>>(
        g16, wt16 + (size_t)(LGNN + l) * DDIM * DDIM, yg, NT);
    ln_kernel<<<(NT + 7) / 8, 256, 0, stream>>>(
        yg, linB + (size_t)l * DDIM, linG + (size_t)l * DDIM, linBe + (size_t)l * DDIM,
        g32, g16, NT);
  }
  // 7. output projection
  outproj_kernel<<<(NT * OUTD + 255) / 256, 256, 0, stream>>>(g32, outW, outB, out);
}
